// Renderer_color_fusion_8787503087841
// MI455X (gfx1250) — matrix operand order repaired and checked by reading, not compiled, not run on hardware
//
#include <hip/hip_runtime.h>
#include <math.h>
#include <stdint.h>

// ---------------------------------------------------------------------------
// MI455X (gfx1250) fused NeRF-style renderer.
// Compute-bound (AI ~1900 FLOP/B): f16 WMMA (v_wmma_f32_16x16x32_f16) for all
// 128-wide GEMMs, activations resident in LDS across layers, one pass over x.
// Weight tiles double-buffered in LDS and streamed with async global->LDS
// loads (ASYNCcnt) via __builtin_amdgcn_global_load_async_to_lds_b128.
// ---------------------------------------------------------------------------

#if defined(__has_builtin)
#  if __has_builtin(__builtin_amdgcn_global_load_async_to_lds_b128)
#    define HAVE_ASYNC_LDS 1
#  else
#    define HAVE_ASYNC_LDS 0
#  endif
#else
#  define HAVE_ASYNC_LDS 0
#endif

typedef _Float16 v16h __attribute__((ext_vector_type(16)));
typedef _Float16 v8h  __attribute__((ext_vector_type(8)));
typedef float    v8f  __attribute__((ext_vector_type(8)));

// Types for the async-to-LDS builtin (signature revealed by clang diagnostics:
// arg0 is `int __vector(4) __device__ *`, i.e. v4i in address space 1).
typedef int v4i __attribute__((vector_size(4 * sizeof(int))));
typedef __attribute__((address_space(1))) v4i* global_v4i_ptr;
typedef __attribute__((address_space(3))) v4i* shared_v4i_ptr;

#define HSTRIDE 200   // halves per h-tile row (192 + 8 pad -> bank-conflict free)

// f16-packed weight offsets inside d_ws (in halves). Row-major [O][Kp].
#define OFF_W0 0          // 128 x 64  (63 -> pad 64)
#define OFF_W1 8192       // 128 x 128
#define OFF_W2 24576
#define OFF_W3 40960
#define OFF_W4 57344
#define OFF_W5 73728      // 128 x 192 (cols 0..62 pts-part, 63 zero, 64..191 h-part)
#define OFF_W6 98304
#define OFF_W7 114688
#define OFF_WB 131072     // 128 x 32  (20 -> pad 32)
#define OFF_WF 135168     // 16  x 128
#define WS_HALVES 137216
#define CPAR_BYTE_OFF (WS_HALVES * 2)   // 274432, 256B aligned

// Const-param block (floats), identical layout in d_ws and LDS.
enum { CP_PB = 0, CP_BB = 1024, CP_BF = 1152, CP_WA = 1168, CP_BA = 1296,
       CP_WQ = 1297, CP_WK = 1745, CP_WV = 2193, CP_WFC = 2641,
       CP_LNG = 3089, CP_LNB = 3117, CP_WRGB = 3145, CP_BRGB = 3229,
       CP_TOTAL = 3232 };

// Dynamic LDS layout (bytes).
#define LDS_WT0   0        // weight buffer 0: 128 x (192+8) halves = 51200 B
#define LDS_WT1   51200    // weight buffer 1: 51200 B
#define LDS_HTILE 102400   // 8 waves x 16 x HSTRIDE halves = 51200 B
#define LDS_FTILE 153600   // 128 x 16 halves = 4096 B
#define LDS_CP    157696   // CP_TOTAL floats = 12928 B
#define LDS_TOTAL 170624   // < 320 KB WGP budget

// ---------------------------------------------------------------------------
// Weight pack: f32 [O][kin_total] -> f16 [O][Kp], with zero padding and the
// layer-5 split remap (k<kin_lo from src k; [kin_lo,pad_to) zero; rest shifted).
// ---------------------------------------------------------------------------
__global__ void pack_w_kernel(const float* __restrict__ src, _Float16* __restrict__ dst,
                              int O, int kin_lo, int pad_to, int kin_total, int Kp) {
  int idx = blockIdx.x * 256 + threadIdx.x;
  if (idx >= O * Kp) return;
  int o = idx / Kp;
  int k = idx - o * Kp;
  float v;
  if (k < kin_lo)      v = src[o * kin_total + k];
  else if (k < pad_to) v = 0.0f;
  else {
    int ks = kin_lo + (k - pad_to);
    v = (ks < kin_total) ? src[o * kin_total + ks] : 0.0f;
  }
  dst[o * Kp + k] = (_Float16)v;
}

// ---------------------------------------------------------------------------
// Fragment loaders (layouts per CDNA5 ISA, wave32).
// A 16x32 f16: lane l -> row l&15; halves [kstart + (l>>4)*8 .. +7] then +16.
// B 32x16 f16: lane l -> col (l&15); halves [kstart + (l>>4)*8 .. +7] then +16.
// ---------------------------------------------------------------------------
static __device__ __forceinline__ v16h load_a_frag(const _Float16* base, int stride,
                                                   int kstart, int lane) {
  const int row = lane & 15;
  const int sel = lane >> 4;
  const _Float16* p0 = base + row * stride + kstart + sel * 8;
  v16h a;
  __builtin_memcpy(&a, p0, 16);
  __builtin_memcpy(reinterpret_cast<char*>(&a) + 16, p0 + 16, 16);
  return a;
}

static __device__ __forceinline__ v16h load_b_frag(const _Float16* w, int stride,
                                                   int cbase, int kstart, int lane) {
  const int n = cbase + (lane & 15);
  const int sel = lane >> 4;
  v16h b;
  const _Float16* q0 = w + n * stride + kstart + sel * 8;
  __builtin_memcpy(&b, q0, 16);
  __builtin_memcpy(reinterpret_cast<char*>(&b) + 16, q0 + 16, 16);
  return b;
}

static __device__ __forceinline__ v8f wmma_f16(v16h a, v16h b, v8f c) {
  return __builtin_amdgcn_wmma_f32_16x16x32_f16(false, a, false, b, (short)0, c,
                                                false, false);
}

// ---------------------------------------------------------------------------
// Cooperative global(f16) -> LDS weight stage, rows padded by 8 halves (+16B)
// so 16 lanes reading 16 different rows of the B tile hit distinct bank quads.
// Async path: GLOBAL_LOAD_ASYNC_TO_LDS_B128 (ASYNCcnt), fully overlapped with
// the previous layer's WMMAs. Fallback: global_load_b128 + ds_store_b128.
// ---------------------------------------------------------------------------
static __device__ __forceinline__ void stage_weights(_Float16* dst, const _Float16* src,
                                                     int O, int Kp, int tid) {
  const int chunks = Kp >> 3;        // 16B chunks per row
  const int stride = Kp + 8;
  const int total  = O * chunks;
  for (int j = tid; j < total; j += 256) {
    int o = j / chunks;
    int c = j - o * chunks;
    _Float16* d = dst + o * stride + (c << 3);
    const _Float16* s = src + o * Kp + (c << 3);
#if HAVE_ASYNC_LDS
    __builtin_amdgcn_global_load_async_to_lds_b128(
        (global_v4i_ptr)(uintptr_t)s,
        (shared_v4i_ptr)(uint32_t)(uintptr_t)d,
        0, 0);
#else
    __builtin_memcpy(d, s, 16);
#endif
  }
}

static __device__ __forceinline__ void wait_weights() {
#if HAVE_ASYNC_LDS
  asm volatile("s_wait_asynccnt 0x0" ::: "memory");
#else
  asm volatile("s_wait_dscnt 0x0" ::: "memory");
#endif
}

// ---------------------------------------------------------------------------
// Main fused kernel: 256 threads (8 wave32), 128 samples per block.
// ---------------------------------------------------------------------------
__global__ __launch_bounds__(256, 1) void fused_nerf_kernel(
    const float* __restrict__ x, const _Float16* __restrict__ wpk,
    const float* __restrict__ cpar, float* __restrict__ out) {
  extern __shared__ char smem[];
  _Float16* wt0   = (_Float16*)(smem + LDS_WT0);
  _Float16* wt1   = (_Float16*)(smem + LDS_WT1);
  _Float16* htile = (_Float16*)(smem + LDS_HTILE);
  _Float16* ftile = (_Float16*)(smem + LDS_FTILE);
  float*    cp    = (float*)(smem + LDS_CP);

  const int tid  = (int)threadIdx.x;
  const int lane = tid & 31;
  const int wave = tid >> 5;
  const int nl   = lane & 15;
  const int sel  = lane >> 4;
  const int s0blk = (int)blockIdx.x * 128;
  const int s0    = s0blk + wave * 16;
  _Float16* htw = htile + wave * (16 * HSTRIDE);

  // Kick off async staging of the gate weights (wt0) and layer-0 weights (wt1).
  stage_weights(wt0, wpk + OFF_WB, 128, 32, tid);
  stage_weights(wt1, wpk + OFF_W0, 128, 64, tid);

  // Const params -> LDS.
  for (int j = tid; j < CP_TOTAL; j += 256) cp[j] = cpar[j];

  // Stage this wave's 16 input rows: pts (k 0..62, pad 63) at h-tile cols 0..63,
  // feats (20, pad 32) temporarily at cols 64..95 for the gate GEMM.
  for (int i = lane; i < 16 * 64; i += 32) {
    int r = i >> 6, k = i & 63;
    float v = (k < 63) ? x[(size_t)(s0 + r) * 110 + k] : 0.0f;
    htw[r * HSTRIDE + k] = (_Float16)v;
  }
  for (int i = lane; i < 16 * 32; i += 32) {
    int r = i >> 5, k = i & 31;
    float v = (k < 20) ? x[(size_t)(s0 + r) * 110 + 63 + k] : 0.0f;
    htw[r * HSTRIDE + 64 + k] = (_Float16)v;
  }
  wait_weights();
  __syncthreads();

  // Gate = feats @ Wb.T + bb, kept packed-f16 in registers in C-fragment layout.
  v8h gate[8];
  {
    const v8f vz = {0, 0, 0, 0, 0, 0, 0, 0};
    v8f g[8];
#pragma unroll
    for (int c = 0; c < 8; ++c) g[c] = vz;
    v16h a = load_a_frag(htw, HSTRIDE, 64, lane);
#pragma unroll
    for (int c = 0; c < 8; ++c) {
      v16h b = load_b_frag(wt0, 32 + 8, c * 16, 0, lane);
      g[c] = wmma_f16(a, b, g[c]);
    }
#pragma unroll
    for (int c = 0; c < 8; ++c) {
      float bc = cp[CP_BB + c * 16 + nl];
#pragma unroll
      for (int r = 0; r < 8; ++r) gate[c][r] = (_Float16)(g[c][r] + bc);
    }
  }
  __syncthreads();   // all waves done reading wt0 before it is restaged

  const int OFFS[8] = {OFF_W0, OFF_W1, OFF_W2, OFF_W3, OFF_W4, OFF_W5, OFF_W6, OFF_W7};
  const int KPS[8]  = {64, 128, 128, 128, 128, 192, 128, 128};

  // Layer li computes from buffer (li&1 ? wt0 : wt1) while the next layer's
  // weights stream asynchronously into the other buffer.
  for (int li = 0; li < 8; ++li) {
    _Float16* cur = (li & 1) ? wt0 : wt1;
    _Float16* nxt = (li & 1) ? wt1 : wt0;
    if (li < 7) stage_weights(nxt, wpk + OFFS[li + 1], 128, KPS[li + 1], tid);
    else        stage_weights(nxt, wpk + OFF_WF, 16, 128, tid);   // feature head

    const int nk      = KPS[li] >> 5;
    const int akbase  = (li == 0 || li == 5) ? 0 : 64;  // layer 0: pts; layer 5: [pts|h]
    const int wstride = KPS[li] + 8;

    const v8f vz = {0, 0, 0, 0, 0, 0, 0, 0};
    v8f acc[8];
#pragma unroll
    for (int c = 0; c < 8; ++c) acc[c] = vz;

    for (int s = 0; s < nk; ++s) {
      v16h a = load_a_frag(htw, HSTRIDE, akbase + s * 32, lane);
#pragma unroll
      for (int c = 0; c < 8; ++c) {
        v16h b = load_b_frag(cur, wstride, c * 16, s * 32, lane);
        acc[c] = wmma_f16(a, b, acc[c]);
      }
    }

    // h = relu((acc + b) * gate); write back f16 into h-tile cols 64..191.
    const float* bptr = cp + CP_PB + li * 128;
#pragma unroll
    for (int c = 0; c < 8; ++c) {
      float bc = bptr[c * 16 + nl];
#pragma unroll
      for (int r = 0; r < 8; ++r) {
        float v = (acc[c][r] + bc) * (float)gate[c][r];
        v = fmaxf(v, 0.0f);
        htw[(sel * 8 + r) * HSTRIDE + 64 + c * 16 + nl] = (_Float16)v;
      }
    }
    asm volatile("s_wait_dscnt 0" ::: "memory");  // own-wave LDS RAW across layers
    wait_weights();                               // next layer's weights have landed
    __syncthreads();                              // and everyone is done with `cur`
  }

  // alpha = relu(h . Wa + ba), one sample per lane (lanes 0..15).
  if (lane < 16) {
    float sum = cp[CP_BA];
    const _Float16* hr = htw + lane * HSTRIDE + 64;
    for (int k = 0; k < 128; ++k) sum += (float)hr[k] * cp[CP_WA + k];
    out[(size_t)(s0 + lane) * 4 + 3] = fmaxf(sum, 0.0f);
  }

  // feature = relu(h @ Wf.T + bf): one 16-col tile, 4 K-steps; Wf is in wt1
  // (staged as "layer 8" during the last MLP layer).
  {
    const v8f vz = {0, 0, 0, 0, 0, 0, 0, 0};
    v8f acc = vz;
#pragma unroll
    for (int s = 0; s < 4; ++s) {
      v16h a = load_a_frag(htw, HSTRIDE, 64 + s * 32, lane);
      v16h b = load_b_frag(wt1, 136, 0, s * 32, lane);
      acc = wmma_f16(a, b, acc);
    }
#pragma unroll
    for (int r = 0; r < 8; ++r) {
      float f = fmaxf(acc[r] + cp[CP_BF + nl], 0.0f);
      ftile[(wave * 16 + sel * 8 + r) * 16 + nl] = (_Float16)f;
    }
  }
  __syncthreads();

  // --------------------------------------------------------------------
  // Tiny 3-token / 4-head attention + layernorm + RGB head: VALU, 1 thr/sample.
  // --------------------------------------------------------------------
  if (tid < 128) {
    const int s = s0blk + tid;
    const float* xr = x + (size_t)s * 110;
    float f[16];
#pragma unroll
    for (int j = 0; j < 16; ++j) f[j] = (float)ftile[tid * 16 + j];

    float kk[3][16], vv[3][16], msk[3];
    for (int t = 0; t < 3; ++t) {
      float hh[28];
#pragma unroll
      for (int j = 0; j < 16; ++j) hh[j] = f[j];
#pragma unroll
      for (int j = 0; j < 9; ++j) hh[16 + j] = xr[83 + t * 9 + j];
#pragma unroll
      for (int j = 0; j < 3; ++j) hh[25 + j] = xr[71 + t * 4 + j];
      msk[t] = xr[71 + t * 4 + 3];
      for (int o = 0; o < 16; ++o) {
        float dk = 0.f, dv = 0.f;
        for (int j = 0; j < 28; ++j) {
          dk += cp[CP_WK + o * 28 + j] * hh[j];
          dv += cp[CP_WV + o * 28 + j] * hh[j];
        }
        kk[t][o] = dk;
        vv[t][o] = dv;
      }
    }

    float rgb[3] = {0.f, 0.f, 0.f};
    for (int qt = 0; qt < 3; ++qt) {
      float hh[28];
#pragma unroll
      for (int j = 0; j < 16; ++j) hh[j] = f[j];
#pragma unroll
      for (int j = 0; j < 9; ++j) hh[16 + j] = xr[83 + qt * 9 + j];
#pragma unroll
      for (int j = 0; j < 3; ++j) hh[25 + j] = xr[71 + qt * 4 + j];

      float q[16];
      for (int o = 0; o < 16; ++o) {
        float d = 0.f;
        for (int j = 0; j < 28; ++j) d += cp[CP_WQ + o * 28 + j] * hh[j];
        q[o] = d;
      }

      float ovec[16];
      for (int hd = 0; hd < 4; ++hd) {
        float lg[3];
        for (int kt = 0; kt < 3; ++kt) {
          float d = 0.f;
#pragma unroll
          for (int j = 0; j < 4; ++j) d += q[hd * 4 + j] * kk[kt][hd * 4 + j];
          d *= 0.5f;  // / sqrt(DK=4)
          lg[kt] = (msk[kt] == 0.0f) ? -1.0e9f : d;
        }
        float mx = fmaxf(lg[0], fmaxf(lg[1], lg[2]));
        float e0 = expf(lg[0] - mx), e1 = expf(lg[1] - mx), e2 = expf(lg[2] - mx);
        float rs = 1.0f / (e0 + e1 + e2);
#pragma unroll
        for (int j = 0; j < 4; ++j)
          ovec[hd * 4 + j] = (e0 * vv[0][hd * 4 + j] + e1 * vv[1][hd * 4 + j] +
                              e2 * vv[2][hd * 4 + j]) * rs;
      }

      float o2[28];
      float mu = 0.f;
      for (int j = 0; j < 28; ++j) {
        float d = 0.f;
        for (int i2 = 0; i2 < 16; ++i2) d += cp[CP_WFC + j * 16 + i2] * ovec[i2];
        o2[j] = d + hh[j];
        mu += o2[j];
      }
      mu *= (1.0f / 28.0f);
      float var = 0.f;
      for (int j = 0; j < 28; ++j) { float d = o2[j] - mu; var += d * d; }
      var *= (1.0f / 28.0f);
      float rinv = rsqrtf(var + 1e-6f);
      for (int j = 0; j < 28; ++j)
        o2[j] = (o2[j] - mu) * rinv * cp[CP_LNG + j] + cp[CP_LNB + j];

      for (int cc = 0; cc < 3; ++cc) {
        float d = cp[CP_BRGB + cc];
        for (int j = 0; j < 28; ++j) d += cp[CP_WRGB + cc * 28 + j] * o2[j];
        rgb[cc] += 1.0f / (1.0f + expf(-d));
      }
    }
    out[(size_t)s * 4 + 0] = rgb[0];
    out[(size_t)s * 4 + 1] = rgb[1];
    out[(size_t)s * 4 + 2] = rgb[2];
  }
}

// ---------------------------------------------------------------------------
// Host launcher. d_in order: x, pts_W[0..7], pts_b[0..7], Wb, bb, Wf, bf,
// Wa, ba, Wq, Wk, Wv, Wfc, ln_g, ln_b, Wrgb, brgb.
// ---------------------------------------------------------------------------
extern "C" void kernel_launch(void* const* d_in, const int* in_sizes, int n_in,
                              void* d_out, int out_size, void* d_ws, size_t ws_size,
                              hipStream_t stream) {
  (void)in_sizes; (void)n_in; (void)out_size; (void)ws_size;

  const float* x = (const float*)d_in[0];
  _Float16* wpk = (_Float16*)d_ws;
  float* cw = (float*)((char*)d_ws + CPAR_BYTE_OFF);

  auto packW = [&](int in_idx, int off, int O, int kin_lo, int pad_to,
                   int kin_total, int Kp) {
    int count = O * Kp;
    pack_w_kernel<<<(count + 255) / 256, 256, 0, stream>>>(
        (const float*)d_in[in_idx], wpk + off, O, kin_lo, pad_to, kin_total, Kp);
  };
  packW(1, OFF_W0, 128, 63, 64, 63, 64);
  packW(2, OFF_W1, 128, 128, 128, 128, 128);
  packW(3, OFF_W2, 128, 128, 128, 128, 128);
  packW(4, OFF_W3, 128, 128, 128, 128, 128);
  packW(5, OFF_W4, 128, 128, 128, 128, 128);
  packW(6, OFF_W5, 128, 63, 64, 191, 192);   // [pts|pad|h] remap for the concat
  packW(7, OFF_W6, 128, 128, 128, 128, 128);
  packW(8, OFF_W7, 128, 128, 128, 128, 128);
  packW(17, OFF_WB, 128, 20, 32, 20, 32);
  packW(19, OFF_WF, 16, 128, 128, 128, 128);

  auto cpy = [&](int dstOff, int in_idx, int nfl) {
    (void)hipMemcpyAsync(cw + dstOff, d_in[in_idx], (size_t)nfl * sizeof(float),
                         hipMemcpyDeviceToDevice, stream);
  };
  for (int i = 0; i < 8; ++i) cpy(CP_PB + i * 128, 9 + i, 128);
  cpy(CP_BB, 18, 128);
  cpy(CP_BF, 20, 16);
  cpy(CP_WA, 21, 128);
  cpy(CP_BA, 22, 1);
  cpy(CP_WQ, 23, 448);
  cpy(CP_WK, 24, 448);
  cpy(CP_WV, 25, 448);
  cpy(CP_WFC, 26, 448);
  cpy(CP_LNG, 27, 28);
  cpy(CP_LNB, 28, 28);
  cpy(CP_WRGB, 29, 84);
  cpy(CP_BRGB, 30, 3);

  (void)hipFuncSetAttribute((const void*)fused_nerf_kernel,
                            hipFuncAttributeMaxDynamicSharedMemorySize, LDS_TOTAL);

  // 524288 samples / 128 per block = 4096 blocks.
  fused_nerf_kernel<<<4096, 256, LDS_TOTAL, stream>>>(x, wpk, cw, (float*)d_out);
}
